// GraphEC_pH_8383776162383
// MI455X (gfx1250) — matrix operand order repaired and checked by reading, not compiled, not run on hardware
//
#include <hip/hip_runtime.h>
#include <hip/hip_bf16.h>
#include <cstring>

// ---------------------------------------------------------------------------
// CDNA5 (gfx1250) graph-transformer forward pass.
// All GEMMs run on the matrix pipe via v_wmma_f32_16x16x32_f16 (f16 in, f32
// accumulate). One wave computes one 16x16 C tile; K is consumed 32/step.
// M and N are multiples of 16 for every GEMM in this network, and all concat
// segment boundaries (128/256/1024) are multiples of 32, so the main K loop
// runs with NO bounds checks and vectorized (float4) A loads; only a ragged
// K tail (K=1036, K=16) takes a guarded scalar path.
// Input flattening assumption: dict insertion order ->
//   0:X 1:h_V 2:edge_index 3:seq 4:batch_id 5..24: top-level params,
//   25+25*L: per-layer params (Wq,bq,Wk,bk,Wv,bv,Wed,ln0g,ln0b,ln1g,ln1b,
//            Wf1,bf1,Wf2,bf2,W11,b11,W12,b12,bng,bnb,Wg1,bg1,Wg2,bg2).
// Workspace requirement: ~600 MB (3 x E x 128 f32 edge buffers dominate).
// ---------------------------------------------------------------------------

#define HID 128
#define HEADS 4
#define RBF_N 16
#define NGRAPH 16

typedef __attribute__((ext_vector_type(16))) _Float16 v16h;
typedef __attribute__((ext_vector_type(8)))  float    v8f;

// ---------------- activation ----------------
__device__ __forceinline__ float act_apply(float x, int act) {
  switch (act) {
    case 1: return x > 0.f ? x : 0.f;                                   // relu
    case 2: return 0.5f * x * (1.f + erff(x * 0.70710678118654752f));   // gelu exact
    case 3: return tanhf(x);                                            // tanh
    case 4: return 1.f / (1.f + expf(-x));                              // sigmoid
    case 5: return x > 0.f ? x : expm1f(x);                             // elu
    default: return x;
  }
}

__device__ __forceinline__ float4 ffma4(float4 a, float4 s, float4 t) {
  float4 r;
  r.x = fmaf(a.x, s.x, t.x); r.y = fmaf(a.y, s.y, t.y);
  r.z = fmaf(a.z, s.z, t.z); r.w = fmaf(a.w, s.w, t.w);
  return r;
}

__device__ __forceinline__ void pack8(v16h& f, int base, float4 a, float4 b) {
  f[base + 0] = (_Float16)a.x; f[base + 1] = (_Float16)a.y;
  f[base + 2] = (_Float16)a.z; f[base + 3] = (_Float16)a.w;
  f[base + 4] = (_Float16)b.x; f[base + 5] = (_Float16)b.y;
  f[base + 6] = (_Float16)b.z; f[base + 7] = (_Float16)b.w;
}

// ---------------- generic WMMA GEMM ----------------
// C[M,N] = act( fetchA(M,K) @ W[K,N] + bias ),  M % 16 == 0, N % 16 == 0
// AMODE 0: A[row*K+k]
// AMODE 1: A[row*K+k]*ascale[k]+ashift[k]                 (fused batch-norm)
// AMODE 2: k<1024 ? A[row*1024+k] : aux0[row*12+k-1024]   (h_V || X concat)
// AMODE 3: concat(aux0[src[row]], aux1[row], aux0[dst[row]]), K=384
template <int AMODE>
__global__ void gemm_wmma(const float* __restrict__ A, int M, int K,
                          const float* __restrict__ W, int N,
                          const float* __restrict__ bias,
                          float* __restrict__ C,
                          const float* __restrict__ ascale,
                          const float* __restrict__ ashift,
                          const float* __restrict__ aux0,
                          const float* __restrict__ aux1,
                          const int* __restrict__ gsrc,
                          const int* __restrict__ gdst,
                          int act) {
  const int lane = threadIdx.x & 31;
  const int wave = threadIdx.x >> 5;
  const int tiles_n = N >> 4;
  const int tiles_m = M >> 4;
  const long tile = (long)blockIdx.x * (blockDim.x >> 5) + wave;
  if (tile >= (long)tiles_m * tiles_n) return;   // whole wave exits together

  const int tm  = (int)(tile / tiles_n);
  const int tn  = (int)(tile % tiles_n);
  const int n_g = tn * 16 + (lane & 15);         // B/C column for this lane
  const int row = tm * 16 + (lane & 15);         // A row for this lane
  const int kbA = (lane >> 4) * 8;               // A K-half base (0 or 8)
  const int kbB = (lane >> 4) * 16;              // unused; B also uses kbA (0 or 8)

  long asrc = 0, adst = 0;
  if (AMODE == 3) { asrc = gsrc[row]; adst = gdst[row]; }

  v8f acc = {};
  int k0 = 0;
  // ---- main loop: full 32-wide chunks, unconditional vector loads ----
  for (; k0 + 32 <= K; k0 += 32) {
    const float* ap;
    if (AMODE == 3) {               // chunk never crosses a 128-boundary
      if (k0 < 128)      ap = aux0 + asrc * HID + k0;
      else if (k0 < 256) ap = aux1 + (long)row * HID + (k0 - 128);
      else               ap = aux0 + adst * HID + (k0 - 256);
    } else if (AMODE == 2) {        // full chunks all lie below 1024
      ap = A + (long)row * 1024 + k0;
    } else {
      ap = A + (long)row * K + k0;
    }
    float4 x0 = *(const float4*)(ap + kbA);
    float4 x1 = *(const float4*)(ap + kbA + 4);
    float4 x2 = *(const float4*)(ap + kbA + 16);
    float4 x3 = *(const float4*)(ap + kbA + 20);
    if (AMODE == 1) {
      const float* sp = ascale + k0 + kbA;
      const float* tp = ashift + k0 + kbA;
      x0 = ffma4(x0, *(const float4*)sp,        *(const float4*)tp);
      x1 = ffma4(x1, *(const float4*)(sp + 4),  *(const float4*)(tp + 4));
      x2 = ffma4(x2, *(const float4*)(sp + 16), *(const float4*)(tp + 16));
      x3 = ffma4(x3, *(const float4*)(sp + 20), *(const float4*)(tp + 20));
    }
    v16h af, bf;
    pack8(af, 0, x0, x1);
    pack8(af, 8, x2, x3);
    // B: lane = column n_g; lanes 0-15 hold K k0+0..7,16..23, lanes 16-31 k0+8..15,24..31
    const float* wp = W + (long)(k0 + kbA) * N + n_g;
#pragma unroll
    for (int e = 0; e < 16; ++e) bf[e] = (_Float16)wp[(long)((e < 8) ? e : (e + 8)) * N];
    acc = __builtin_amdgcn_wmma_f32_16x16x32_f16(
        false, af, false, bf, (short)0, acc, false, false);
  }
  // ---- ragged K tail (K=1036 or K=16): guarded scalar path ----
  if (k0 < K) {
    v16h af, bf;
#pragma unroll
    for (int e = 0; e < 16; ++e) {
      const int kr = (e < 8) ? (kbA + e) : (kbA + 8 + e);
      const int k = k0 + kr;
      float x = 0.f;
      if (k < K) {
        if (AMODE == 0)      x = A[(long)row * K + k];
        else if (AMODE == 1) x = A[(long)row * K + k] * ascale[k] + ashift[k];
        else if (AMODE == 2) x = (k < 1024) ? A[(long)row * 1024 + k]
                                            : aux0[(long)row * 12 + (k - 1024)];
        else {
          if (k < 128)      x = aux0[asrc * HID + k];
          else if (k < 256) x = aux1[(long)row * HID + (k - 128)];
          else              x = aux0[adst * HID + (k - 256)];
        }
      }
      af[e] = (_Float16)x;
    }
#pragma unroll
    for (int e = 0; e < 16; ++e) {
      const int k = k0 + ((e < 8) ? (kbA + e) : (kbA + 8 + e));
      bf[e] = (_Float16)((k < K) ? W[(long)k * N + n_g] : 0.f);
    }
    acc = __builtin_amdgcn_wmma_f32_16x16x32_f16(
        false, af, false, bf, (short)0, acc, false, false);
  }

  const float bval = (bias != nullptr) ? bias[n_g] : 0.f;
  const int m0 = tm * 16 + ((lane >> 4) * 8);     // C: VGPR i -> M=i (+8 hi lanes)
#pragma unroll
  for (int i = 0; i < 8; ++i)
    C[(long)(m0 + i) * N + n_g] = act_apply(acc[i] + bval, act);
}

// ---------------- elementwise / reduction kernels ----------------
__device__ __forceinline__ unsigned f2o(float f) {
  unsigned b = __float_as_uint(f);
  return (b & 0x80000000u) ? ~b : (b | 0x80000000u);
}
__device__ __forceinline__ float o2f(unsigned u) {
  unsigned b = (u & 0x80000000u) ? (u & 0x7FFFFFFFu) : ~u;
  return __uint_as_float(b);
}

__global__ void k_fill_u32(unsigned* p, long n, unsigned v) {
  for (long i = (long)blockIdx.x * blockDim.x + threadIdx.x; i < n;
       i += (long)gridDim.x * blockDim.x) p[i] = v;
}

__global__ void k_rbf(const float* __restrict__ X, const int* __restrict__ src,
                      const int* __restrict__ dst, long E, float* __restrict__ out) {
  long e = (long)blockIdx.x * blockDim.x + threadIdx.x;
  if (e >= E) return;
  const long s = src[e], d = dst[e];
  float acc = 1e-12f;
#pragma unroll
  for (int j = 0; j < 3; ++j) {
    float df = X[d * 12 + 3 + j] - X[s * 12 + 3 + j];   // X[:,1,:] = CA atom
    acc += df * df;
  }
  const float dist = sqrtf(acc);
#pragma unroll
  for (int i = 0; i < RBF_N; ++i) {
    const float c = (20.f / 15.f) * i;                  // linspace(0,20,16)
    const float t = (dist - c) * (1.f / 1.25f);         // / (20/16)
    out[e * RBF_N + i] = expf(-t * t);
  }
}

__global__ void k_colstats(const float* __restrict__ X, const float* __restrict__ Y,
                           long M, int C, float* __restrict__ sum,
                           float* __restrict__ sumsq) {
  const int c = threadIdx.x;
  if (c >= C) return;
  float s = 0.f, q = 0.f;
  if (Y) {
    for (long r = blockIdx.x; r < M; r += gridDim.x) {
      const float x = X[r * C + c] + Y[r * C + c];
      s += x; q += x * x;
    }
  } else {
    for (long r = blockIdx.x; r < M; r += gridDim.x) {
      const float x = X[r * C + c];
      s += x; q += x * x;
    }
  }
  atomicAdd(&sum[c], s);
  atomicAdd(&sumsq[c], q);
}

__global__ void k_bn_finalize(const float* sum, const float* sumsq, long M,
                              const float* g, const float* b, float eps,
                              float* scale, float* shift) {
  const int c = threadIdx.x;
  const float mu = sum[c] / (float)M;
  const float var = sumsq[c] / (float)M - mu * mu;
  const float s = g[c] * rsqrtf(var + eps);
  scale[c] = s;
  shift[c] = b[c] - mu * s;
}

__global__ void k_bn_apply_add(float* __restrict__ X, const float* __restrict__ Y,
                               const float* __restrict__ scale,
                               const float* __restrict__ shift, long n, int C) {
  for (long i = (long)blockIdx.x * blockDim.x + threadIdx.x; i < n;
       i += (long)gridDim.x * blockDim.x) {
    const int c = (int)(i % C);
    X[i] = (X[i] + Y[i]) * scale[c] + shift[c];
  }
}

// attention logits: a[e,h] = (1/sqrt(32)) * sum_d q[dst][h,d]*(k[src][h,d]+e[h,d])
__global__ void k_attn(const float* __restrict__ q, const float* __restrict__ kk,
                       const float* __restrict__ ep, const int* __restrict__ src,
                       const int* __restrict__ dst, long E, float* __restrict__ a) {
  const long e = (long)blockIdx.x * (blockDim.x >> 5) + (threadIdx.x >> 5);
  if (e >= E) return;
  const int lane = threadIdx.x & 31;
  const long s = src[e], d = dst[e];
  float p[HEADS];
#pragma unroll
  for (int h = 0; h < HEADS; ++h) {
    const int c = h * 32 + lane;
    p[h] = q[d * HID + c] * (kk[s * HID + c] + ep[e * HID + c]);
  }
#pragma unroll
  for (int off = 16; off > 0; off >>= 1)
#pragma unroll
    for (int h = 0; h < HEADS; ++h) p[h] += __shfl_xor(p[h], off, 32);
  if (lane == 0)
#pragma unroll
    for (int h = 0; h < HEADS; ++h)
      a[e * HEADS + h] = p[h] * 0.17677669529663687f;
}

__global__ void k_segmax(const float* __restrict__ a, const int* __restrict__ idx,
                         int H, long E, unsigned* __restrict__ mb) {
  const long i = (long)blockIdx.x * blockDim.x + threadIdx.x;
  if (i >= E * H) return;
  atomicMax(&mb[(long)idx[i / H] * H + (int)(i % H)], f2o(a[i]));
}

__global__ void k_segexp(float* __restrict__ a, const int* __restrict__ idx, int H,
                         long E, const unsigned* __restrict__ mb,
                         float* __restrict__ sb) {
  const long i = (long)blockIdx.x * blockDim.x + threadIdx.x;
  if (i >= E * H) return;
  const long seg = (long)idx[i / H] * H + (int)(i % H);
  const float ex = expf(a[i] - o2f(mb[seg]));
  a[i] = ex;
  atomicAdd(&sb[seg], ex);
}

// dh[dst] += (v[src] + e) * alpha
__global__ void k_attn_out(const float* __restrict__ aexp, const float* __restrict__ sb,
                           const float* __restrict__ v, const float* __restrict__ ep,
                           const int* __restrict__ src, const int* __restrict__ dst,
                           long E, float* __restrict__ dh) {
  const long e = (long)blockIdx.x * (blockDim.x >> 5) + (threadIdx.x >> 5);
  if (e >= E) return;
  const int lane = threadIdx.x & 31;
  const long s = src[e], d = dst[e];
  float al[HEADS];
#pragma unroll
  for (int h = 0; h < HEADS; ++h)
    al[h] = aexp[e * HEADS + h] / (sb[d * HEADS + h] + 1e-16f);
#pragma unroll
  for (int h = 0; h < HEADS; ++h) {
    const int c = h * 32 + lane;
    atomicAdd(&dh[d * HID + c], (v[s * HID + c] + ep[e * HID + c]) * al[h]);
  }
}

// out = LayerNorm(base + add) * g + b   (one wave per 128-wide row)
__global__ void k_ln(const float* __restrict__ base, const float* __restrict__ add,
                     const float* __restrict__ g, const float* __restrict__ b,
                     float* __restrict__ out, long Nn) {
  const long row = (long)blockIdx.x * (blockDim.x >> 5) + (threadIdx.x >> 5);
  if (row >= Nn) return;
  const int lane = threadIdx.x & 31;
  float x[4];
#pragma unroll
  for (int j = 0; j < 4; ++j) {
    const int c = lane + j * 32;
    x[j] = base[row * HID + c] + (add ? add[row * HID + c] : 0.f);
  }
  float s = x[0] + x[1] + x[2] + x[3];
#pragma unroll
  for (int off = 16; off > 0; off >>= 1) s += __shfl_xor(s, off, 32);
  const float mu = s * (1.f / HID);
  float vs = 0.f;
#pragma unroll
  for (int j = 0; j < 4; ++j) { const float d = x[j] - mu; vs += d * d; }
#pragma unroll
  for (int off = 16; off > 0; off >>= 1) vs += __shfl_xor(vs, off, 32);
  const float r = rsqrtf(vs * (1.f / HID) + 1e-5f);
#pragma unroll
  for (int j = 0; j < 4; ++j) {
    const int c = lane + j * 32;
    out[row * HID + c] = (x[j] - mu) * r * g[c] + b[c];
  }
}

__global__ void k_graph_accum(const float* __restrict__ hV, const int* __restrict__ bid,
                              long Nn, float* __restrict__ cV, float* __restrict__ cnt) {
  const long row = (long)blockIdx.x * (blockDim.x >> 5) + (threadIdx.x >> 5);
  if (row >= Nn) return;
  const int lane = threadIdx.x & 31;
  const int g = bid[row];
#pragma unroll
  for (int j = 0; j < 4; ++j) {
    const int c = lane + j * 32;
    atomicAdd(&cV[(long)g * HID + c], hV[row * HID + c]);
  }
  if (lane == 0) atomicAdd(&cnt[g], 1.f);
}

__global__ void k_div_rows(float* cV, const float* cnt) {
  cV[(long)blockIdx.x * HID + threadIdx.x] /= cnt[blockIdx.x];
}

__global__ void k_apply_gate(float* __restrict__ hV, const int* __restrict__ bid,
                             const float* __restrict__ gate, long n) {
  for (long i = (long)blockIdx.x * blockDim.x + threadIdx.x; i < n;
       i += (long)gridDim.x * blockDim.x)
    hV[i] *= gate[(long)bid[i / HID] * HID + (int)(i % HID)];
}

__global__ void k_scores4(const float* __restrict__ s16, const float* __restrict__ fc2,
                          const float* __restrict__ fb2, float* __restrict__ s4, long Nn) {
  const long n = (long)blockIdx.x * blockDim.x + threadIdx.x;
  if (n >= Nn) return;
#pragma unroll
  for (int h = 0; h < HEADS; ++h) {
    float acc = fb2[h];
#pragma unroll
    for (int j = 0; j < 16; ++j) acc += s16[n * 16 + j] * fc2[j * HEADS + h];
    s4[n * HEADS + h] = acc;
  }
}

__global__ void k_emb(const float* __restrict__ aexp, const float* __restrict__ sb,
                      const int* __restrict__ bid, const float* __restrict__ hV,
                      long Nn, float* __restrict__ emb) {
  const long row = (long)blockIdx.x * (blockDim.x >> 5) + (threadIdx.x >> 5);
  if (row >= Nn) return;
  const int lane = threadIdx.x & 31;
  const int g = bid[row];
  float w = 0.f;
#pragma unroll
  for (int h = 0; h < HEADS; ++h)
    w += aexp[row * HEADS + h] / (sb[(long)g * HEADS + h] + 1e-16f);
#pragma unroll
  for (int j = 0; j < 4; ++j) {
    const int c = lane + j * 32;
    atomicAdd(&emb[(long)g * HID + c], w * hV[row * HID + c]);
  }
}

__global__ void k_head(const float* __restrict__ t, const float* __restrict__ FC2,
                       const float* __restrict__ FB2, float* __restrict__ out) {
  const int g = threadIdx.x;
  if (g >= NGRAPH) return;
  float l[3];
#pragma unroll
  for (int o = 0; o < 3; ++o) {
    float a = FB2[o];
    for (int k = 0; k < HID; ++k) a += t[(long)g * HID + k] * FC2[k * 3 + o];
    l[o] = a;
  }
  const float m = fmaxf(l[0], fmaxf(l[1], l[2]));
  float es[3], ssum = 0.f;
#pragma unroll
  for (int o = 0; o < 3; ++o) { es[o] = expf(l[o] - m); ssum += es[o]; }
#pragma unroll
  for (int o = 0; o < 3; ++o) out[g * 3 + o] = es[o] / ssum;
}

// ---------------- host-side launcher ----------------
static inline void gemm(hipStream_t s, int amode, const float* A, long M, int K,
                        const float* W, int N, const float* bias, float* C,
                        const float* ascale, const float* ashift,
                        const float* aux0, const float* aux1,
                        const int* gsrc, const int* gdst, int act) {
  const long tiles = (M / 16) * (long)(N / 16);
  const int waves = 8;
  const unsigned blocks = (unsigned)((tiles + waves - 1) / waves);
  dim3 bd(32 * waves);
  switch (amode) {
    case 0: gemm_wmma<0><<<blocks, bd, 0, s>>>(A, (int)M, K, W, N, bias, C, ascale, ashift, aux0, aux1, gsrc, gdst, act); break;
    case 1: gemm_wmma<1><<<blocks, bd, 0, s>>>(A, (int)M, K, W, N, bias, C, ascale, ashift, aux0, aux1, gsrc, gdst, act); break;
    case 2: gemm_wmma<2><<<blocks, bd, 0, s>>>(A, (int)M, K, W, N, bias, C, ascale, ashift, aux0, aux1, gsrc, gdst, act); break;
    default: gemm_wmma<3><<<blocks, bd, 0, s>>>(A, (int)M, K, W, N, bias, C, ascale, ashift, aux0, aux1, gsrc, gdst, act); break;
  }
}

extern "C" void kernel_launch(void* const* d_in, const int* in_sizes, int n_in,
                              void* d_out, int out_size, void* d_ws, size_t ws_size,
                              hipStream_t stream) {
  const float* X   = (const float*)d_in[0];
  const float* hVr = (const float*)d_in[1];
  const int* eidx  = (const int*)d_in[2];
  const int* bid   = (const int*)d_in[4];
  const long N = in_sizes[3];
  const long E = in_sizes[2] / 2;
  const int* src = eidx;
  const int* dst = eidx + E;

  auto F = [&](int i) { return (const float*)d_in[i]; };
  enum { pWn = 5, pbn, pWe, pbe, pbnn_g, pbnn_b, pbne_g, pbne_b, pWv, pbv,
         pWe2, pbe2, pfc1, pfb1, pfc2, pfb2, pFC1, pFB1, pFC2, pFB2, pL0 };
  enum { lWq = 0, lbq, lWk, lbk, lWv, lbv, lWed, lln0g, lln0b, lln1g, lln1b,
         lWf1, lbf1, lWf2, lbf2, lW11, lb11, lW12, lb12, lbng, lbnb,
         lWg1, lbg1, lWg2, lbg2 };

  // ---- workspace carve-out ----
  size_t off = 0;
  auto carve = [&](size_t elems) {
    void* p = (char*)d_ws + off;
    off += ((elems * 4 + 255) / 256) * 256;
    return p;
  };
  float* hVb  = (float*)carve(N * HID);
  float* bufA = (float*)carve(E * HID);   // rbf / eproj / msg1
  float* bufB = (float*)carve(E * HID);   // hE (persistent across layers)
  float* bufC = (float*)carve(E * HID);   // edge scratch / msg2
  float* tno  = (float*)carve(N * HID);
  float* qb   = (float*)carve(N * HID);
  float* kbf  = (float*)carve(N * HID);
  float* vbf  = (float*)carve(N * HID);
  float* dh   = (float*)carve(N * HID);
  float* ffm  = (float*)carve(N * 512);
  float* attn = (float*)carve(E * HEADS);
  unsigned* mb = (unsigned*)carve(N * HEADS);
  float* sb   = (float*)carve(N * HEADS);
  float* sc16 = (float*)carve(N * 16);
  float* sc4  = (float*)carve(N * HEADS);
  float* stats = (float*)carve(512);      // sum|sumsq|scale|shift (128 each)
  float* cV   = (float*)carve(NGRAPH * HID);
  float* cnt  = (float*)carve(NGRAPH);
  float* gate = (float*)carve(NGRAPH * HID);
  float* gmid = (float*)carve(NGRAPH * HID);
  float* emb  = (float*)carve(NGRAPH * HID);
  float* thd  = (float*)carve(NGRAPH * HID);

  float neg = -3.0e38f; unsigned nb_; memcpy(&nb_, &neg, 4);
  const unsigned NEGINF_ORD = ~nb_;   // f2o(-3e38) computed host-side

  const unsigned TPB = 256;
  auto gs1 = [&](long n) { return (unsigned)((n + TPB - 1) / TPB); };
  auto gsw = [&](long n) { return (unsigned)((n + 7) / 8); };  // 8 waves/block

  // ---- stage 1: node embedding (concat GEMM -> BN folded -> GEMM) ----
  hipMemsetAsync(stats, 0, 256 * 4, stream);
  gemm(stream, 2, hVr, N, 1036, F(pWn), HID, F(pbn), tno, 0, 0, X, 0, 0, 0, 0);
  k_colstats<<<1024, 128, 0, stream>>>(tno, nullptr, N, HID, stats, stats + 128);
  k_bn_finalize<<<1, 128, 0, stream>>>(stats, stats + 128, N, F(pbnn_g), F(pbnn_b),
                                       1e-5f, stats + 256, stats + 384);
  gemm(stream, 1, tno, N, HID, F(pWv), HID, F(pbv), hVb, stats + 256, stats + 384,
       0, 0, 0, 0, 0);

  // ---- stage 2: edge embedding ----
  k_rbf<<<gs1(E), TPB, 0, stream>>>(X, src, dst, E, bufA);
  hipMemsetAsync(stats, 0, 256 * 4, stream);
  gemm(stream, 0, bufA, E, RBF_N, F(pWe), HID, F(pbe), bufC, 0, 0, 0, 0, 0, 0, 0);
  k_colstats<<<1024, 128, 0, stream>>>(bufC, nullptr, E, HID, stats, stats + 128);
  k_bn_finalize<<<1, 128, 0, stream>>>(stats, stats + 128, E, F(pbne_g), F(pbne_b),
                                       1e-5f, stats + 256, stats + 384);
  gemm(stream, 1, bufC, E, HID, F(pWe2), HID, F(pbe2), bufB, stats + 256,
       stats + 384, 0, 0, 0, 0, 0);

  // ---- GNN layers ----
  for (int L = 0; L < 2; ++L) {
    const int pb = pL0 + 25 * L;
    gemm(stream, 0, hVb, N, HID, F(pb + lWq), HID, F(pb + lbq), qb, 0,0,0,0,0,0,0);
    gemm(stream, 0, hVb, N, HID, F(pb + lWk), HID, F(pb + lbk), kbf, 0,0,0,0,0,0,0);
    gemm(stream, 0, hVb, N, HID, F(pb + lWv), HID, F(pb + lbv), vbf, 0,0,0,0,0,0,0);
    gemm(stream, 0, bufB, E, HID, F(pb + lWed), HID, nullptr, bufA, 0,0,0,0,0,0,0);
    // attention logits + segment softmax over dst
    k_attn<<<gsw(E), TPB, 0, stream>>>(qb, kbf, bufA, src, dst, E, attn);
    k_fill_u32<<<gs1(N * HEADS), TPB, 0, stream>>>(mb, N * HEADS, NEGINF_ORD);
    hipMemsetAsync(sb, 0, N * HEADS * 4, stream);
    k_segmax<<<gs1(E * HEADS), TPB, 0, stream>>>(attn, dst, HEADS, E, mb);
    k_segexp<<<gs1(E * HEADS), TPB, 0, stream>>>(attn, dst, HEADS, E, mb, sb);
    hipMemsetAsync(dh, 0, N * HID * 4, stream);
    k_attn_out<<<gsw(E), TPB, 0, stream>>>(attn, sb, vbf, bufA, src, dst, E, dh);
    k_ln<<<gsw(N), TPB, 0, stream>>>(hVb, dh, F(pb + lln0g), F(pb + lln0b), hVb, N);
    // FFN + LN
    gemm(stream, 0, hVb, N, HID, F(pb + lWf1), 512, F(pb + lbf1), ffm, 0,0,0,0,0,0, 1);
    gemm(stream, 0, ffm, N, 512, F(pb + lWf2), HID, F(pb + lbf2), tno, 0,0,0,0,0,0, 0);
    k_ln<<<gsw(N), TPB, 0, stream>>>(hVb, tno, F(pb + lln1g), F(pb + lln1b), hVb, N);
    // edge message MLP (gathered concat) + batch-norm update of hE
    gemm(stream, 3, nullptr, E, 3 * HID, F(pb + lW11), HID, F(pb + lb11), bufA,
         0, 0, hVb, bufB, src, dst, 2 /*gelu*/);
    gemm(stream, 0, bufA, E, HID, F(pb + lW12), HID, F(pb + lb12), bufC, 0,0,0,0,0,0, 0);
    hipMemsetAsync(stats, 0, 256 * 4, stream);
    k_colstats<<<1024, 128, 0, stream>>>(bufB, bufC, E, HID, stats, stats + 128);
    k_bn_finalize<<<1, 128, 0, stream>>>(stats, stats + 128, E, F(pb + lbng),
                                         F(pb + lbnb), 1e-5f, stats + 256, stats + 384);
    k_bn_apply_add<<<4096, TPB, 0, stream>>>(bufB, bufC, stats + 256, stats + 384,
                                             E * HID, HID);
    // graph gating
    hipMemsetAsync(cV, 0, NGRAPH * HID * 4, stream);
    hipMemsetAsync(cnt, 0, NGRAPH * 4, stream);
    k_graph_accum<<<gsw(N), TPB, 0, stream>>>(hVb, bid, N, cV, cnt);
    k_div_rows<<<NGRAPH, HID, 0, stream>>>(cV, cnt);
    gemm(stream, 0, cV, NGRAPH, HID, F(pb + lWg1), HID, F(pb + lbg1), gmid, 0,0,0,0,0,0, 1);
    gemm(stream, 0, gmid, NGRAPH, HID, F(pb + lWg2), HID, F(pb + lbg2), gate, 0,0,0,0,0,0, 4);
    k_apply_gate<<<4096, TPB, 0, stream>>>(hVb, bid, gate, N * HID);
  }

  // ---- readout ----
  gemm(stream, 0, hVb, N, HID, F(pfc1), 16, F(pfb1), sc16, 0,0,0,0,0,0, 3 /*tanh*/);
  k_scores4<<<gs1(N), TPB, 0, stream>>>(sc16, F(pfc2), F(pfb2), sc4, N);
  k_fill_u32<<<1, TPB, 0, stream>>>(mb, NGRAPH * HEADS, NEGINF_ORD);
  hipMemsetAsync(sb, 0, NGRAPH * HEADS * 4, stream);
  k_segmax<<<gs1(N * HEADS), TPB, 0, stream>>>(sc4, bid, HEADS, N, mb);
  k_segexp<<<gs1(N * HEADS), TPB, 0, stream>>>(sc4, bid, HEADS, N, mb, sb);
  hipMemsetAsync(emb, 0, NGRAPH * HID * 4, stream);
  k_emb<<<gsw(N), TPB, 0, stream>>>(sc4, sb, bid, hVb, N, emb);
  gemm(stream, 0, emb, NGRAPH, HID, F(pFC1), HID, F(pFB1), thd, 0,0,0,0,0,0, 5 /*elu*/);
  k_head<<<1, NGRAPH, 0, stream>>>(thd, F(pFC2), F(pFB2), (float*)d_out);
  (void)n_in; (void)out_size; (void)ws_size; (void)in_sizes;
}
